// MultiHeadSelfAttention_60395830116573
// MI455X (gfx1250) — compile-verified, matrix operand order checked by reading, not run on hardware
//
#include <hip/hip_runtime.h>

typedef __bf16 bf16_t;
typedef __attribute__((ext_vector_type(16))) __bf16 v16bf;
typedef __attribute__((ext_vector_type(8)))  __bf16 v8bf;
typedef __attribute__((ext_vector_type(4)))  __bf16 v4bf;
typedef __attribute__((ext_vector_type(8)))  float  v8f;

#define L_SEQ   2048
#define NB      2
#define E_DIM   1024
#define H_HEADS 16
#define DH      64
#define NH      (NB * H_HEADS)    // 32 heads total
#define M_ROWS  (L_SEQ * NB)      // 4096
#define SCALE   0.125f            // DH^-0.5
#define NEGV    (-10000.0f)

union FragU { v16bf v; v8bf h[2]; };

// Load a 16x32 bf16 fragment in WMMA A-layout from a row-major matrix.
// lanes 0-15: row = lane, K chunks {k0..k0+7, k0+16..k0+23}
// lanes 16-31: row = lane-16, K chunks {k0+8..k0+15, k0+24..k0+31}
// For B (K x 16), pass the stored-transposed matrix (row = output column):
// identical pattern since W is stored [out,in] row-major.
__device__ __forceinline__ v16bf load_frag_rm(const bf16_t* __restrict__ p,
                                              int ld, int row0, int k0, int lane) {
  int m  = lane & 15;
  int kk = k0 + ((lane & 16) ? 8 : 0);
  const bf16_t* base = p + (size_t)(row0 + m) * ld + kk;
  FragU f;
  f.h[0] = *(const v8bf*)(base);
  f.h[1] = *(const v8bf*)(base + 16);
  return f.v;
}

__device__ __forceinline__ v8f wmma_bf16(v16bf a, v16bf b, v8f c) {
  return __builtin_amdgcn_wmma_f32_16x16x32_bf16(
      /*neg_a=*/false, a, /*neg_b=*/false, b,
      /*c_mod=*/(short)0, c, /*reuse_a=*/false, /*reuse_b=*/false);
}

// ---------------- f32 -> bf16 conversion (vectorized x4) ----------------
__global__ void cvt_f32_bf16_v4(const float* __restrict__ src,
                                bf16_t* __restrict__ dst, int n4) {
  int i = blockIdx.x * blockDim.x + threadIdx.x;
  if (i < n4) {
    float4 f = ((const float4*)src)[i];
    v4bf o;
    o[0] = (bf16_t)f.x; o[1] = (bf16_t)f.y;
    o[2] = (bf16_t)f.z; o[3] = (bf16_t)f.w;
    ((v4bf*)dst)[i] = o;
  }
}

// ---------------- fused QKV projection ----------------
// grid: (M/64, E/64, 3); block: 128 (4 waves, 16 rows x 64 cols each)
// mode 0: Q -> [head][L][DH] bf16, pre-scaled by SCALE
// mode 1: K -> [head][L][DH] bf16
// mode 2: V -> [head][DH][L] bf16 (transposed for PV B-fragments)
__global__ __launch_bounds__(128)
void qkv_gemm(const bf16_t* __restrict__ X,
              const bf16_t* __restrict__ Wq, const bf16_t* __restrict__ Wk,
              const bf16_t* __restrict__ Wv,
              const float* __restrict__ bq, const float* __restrict__ bk,
              const float* __restrict__ bv,
              bf16_t* __restrict__ Qh, bf16_t* __restrict__ Kh,
              bf16_t* __restrict__ Vt) {
  const int lane = threadIdx.x & 31;
  const int wave = threadIdx.x >> 5;
  const int mode = blockIdx.z;
  const bf16_t* W   = (mode == 0) ? Wq : ((mode == 1) ? Wk : Wv);
  const float* bias = (mode == 0) ? bq : ((mode == 1) ? bk : bv);
  const int rbase = blockIdx.x * 64 + wave * 16;
  const int cbase = blockIdx.y * 64;

  v8f acc[4] = {};
  for (int k0 = 0; k0 < E_DIM; k0 += 32) {
    v16bf a = load_frag_rm(X, E_DIM, rbase, k0, lane);
#pragma unroll
    for (int t = 0; t < 4; ++t) {
      v16bf b = load_frag_rm(W, E_DIM, cbase + t * 16, k0, lane);
      acc[t] = wmma_bf16(a, b, acc[t]);
    }
  }

  const int n    = lane & 15;
  const int moff = (lane & 16) ? 8 : 0;
#pragma unroll
  for (int t = 0; t < 4; ++t) {
    int c = cbase + t * 16 + n;
    float bc = bias[c];
    int h = c >> 6, d = c & 63;
#pragma unroll
    for (int v = 0; v < 8; ++v) {
      int r  = rbase + moff + v;
      int l  = r >> 1;          // sequence position (row-major [L][N][E])
      int nb = r & 1;           // batch index
      int bh = nb * H_HEADS + h;
      float val = acc[t][v] + bc;
      if (mode == 0) {
        Qh[((size_t)bh * L_SEQ + l) * DH + d] = (bf16_t)(val * SCALE);
      } else if (mode == 1) {
        Kh[((size_t)bh * L_SEQ + l) * DH + d] = (bf16_t)val;
      } else {
        Vt[((size_t)bh * DH + d) * L_SEQ + l] = (bf16_t)val;
      }
    }
  }
}

// ---------------- causal flash attention ----------------
// grid: (NH, L/64); block: 128 (4 independent waves, 16 q-rows each)
__global__ __launch_bounds__(128)
void attn_kernel(const bf16_t* __restrict__ Qh, const bf16_t* __restrict__ Kh,
                 const bf16_t* __restrict__ Vt, bf16_t* __restrict__ Ao) {
  __shared__ alignas(16) bf16_t pbuf[4 * 16 * 32];  // per-wave P staging
  const int lane = threadIdx.x & 31;
  const int wave = threadIdx.x >> 5;
  const int head = blockIdx.x;
  const int q0   = blockIdx.y * 64 + wave * 16;

  const bf16_t* Q = Qh + (size_t)head * L_SEQ * DH;
  const bf16_t* K = Kh + (size_t)head * L_SEQ * DH;
  const bf16_t* V = Vt + (size_t)head * DH * L_SEQ;
  bf16_t* pw = pbuf + wave * (16 * 32);

  // resident Q fragments (16 rows x 64 = two k-steps)
  v16bf qf0 = load_frag_rm(Q, DH, q0, 0, lane);
  v16bf qf1 = load_frag_rm(Q, DH, q0, 32, lane);

  v8f acc[4] = {};
  float rmax[8], rsum[8];
#pragma unroll
  for (int v = 0; v < 8; ++v) { rmax[v] = -3.0e38f; rsum[v] = 0.0f; }

  const int n    = lane & 15;
  const int moff = (lane & 16) ? 8 : 0;
  // causal: this wave only needs keys <= q0+15
  const int nkb = (q0 + 16 + 31) >> 5;

  for (int kb = 0; kb < nkb; ++kb) {
    const int kc = kb * 32;
    // S = Q K^T : two 16x16 tiles over 32 keys
    v8f s0 = {}, s1 = {};
    {
      v16bf b00 = load_frag_rm(K, DH, kc, 0, lane);
      v16bf b01 = load_frag_rm(K, DH, kc, 32, lane);
      s0 = wmma_bf16(qf0, b00, s0);
      s0 = wmma_bf16(qf1, b01, s0);
      v16bf b10 = load_frag_rm(K, DH, kc + 16, 0, lane);
      v16bf b11 = load_frag_rm(K, DH, kc + 16, 32, lane);
      s1 = wmma_bf16(qf0, b10, s1);
      s1 = wmma_bf16(qf1, b11, s1);
    }
    // causal mask (reference uses -10000, not -inf)
#pragma unroll
    for (int v = 0; v < 8; ++v) {
      int qg = q0 + moff + v;
      if (kc + n > qg)      s0[v] = NEGV;
      if (kc + 16 + n > qg) s1[v] = NEGV;
    }
    // online softmax; rows live across the 16-lane half-waves
    float alpha[8];
#pragma unroll
    for (int v = 0; v < 8; ++v) {
      float mx = fmaxf(s0[v], s1[v]);
      mx = fmaxf(mx, __shfl_xor(mx, 1));
      mx = fmaxf(mx, __shfl_xor(mx, 2));
      mx = fmaxf(mx, __shfl_xor(mx, 4));
      mx = fmaxf(mx, __shfl_xor(mx, 8));
      float mnew = fmaxf(rmax[v], mx);
      alpha[v] = __expf(rmax[v] - mnew);
      rmax[v] = mnew;
      float p0 = __expf(s0[v] - mnew);
      float p1 = __expf(s1[v] - mnew);
      s0[v] = p0; s1[v] = p1;
      float ps = p0 + p1;
      ps += __shfl_xor(ps, 1);
      ps += __shfl_xor(ps, 2);
      ps += __shfl_xor(ps, 4);
      ps += __shfl_xor(ps, 8);
      rsum[v] = rsum[v] * alpha[v] + ps;
    }
#pragma unroll
    for (int t = 0; t < 4; ++t)
#pragma unroll
      for (int v = 0; v < 8; ++v)
        acc[t][v] *= alpha[v];

    // C-layout P -> LDS -> A-fragment layout
#pragma unroll
    for (int v = 0; v < 8; ++v) {
      int m = moff + v;
      pw[m * 32 + n]      = (bf16_t)s0[v];
      pw[m * 32 + 16 + n] = (bf16_t)s1[v];
    }
    asm volatile("s_wait_dscnt 0" ::: "memory");
    v16bf pf;
    {
      int mm = lane & 15;
      int kk = (lane & 16) ? 8 : 0;
      FragU f;
      f.h[0] = *(const v8bf*)(pw + mm * 32 + kk);
      f.h[1] = *(const v8bf*)(pw + mm * 32 + kk + 16);
      pf = f.v;
    }
    // O += P * V  (B-fragments from transposed V, rows = d)
#pragma unroll
    for (int t = 0; t < 4; ++t) {
      v16bf vb = load_frag_rm(V, L_SEQ, t * 16, kc, lane);
      acc[t] = wmma_bf16(pf, vb, acc[t]);
    }
  }

#pragma unroll
  for (int v = 0; v < 8; ++v) rsum[v] = 1.0f / rsum[v];
  const int nb = head >> 4;        // head = batch*H + h
  const int h  = head & 15;
#pragma unroll
  for (int t = 0; t < 4; ++t) {
#pragma unroll
    for (int v = 0; v < 8; ++v) {
      int qg = q0 + moff + v;
      int r  = qg * NB + nb;
      int c  = h * DH + t * 16 + n;
      Ao[(size_t)r * E_DIM + c] = (bf16_t)(acc[t][v] * rsum[v]);
    }
  }
}

// ---------------- output projection (f32 result) ----------------
__global__ __launch_bounds__(128)
void out_gemm(const bf16_t* __restrict__ A, const bf16_t* __restrict__ Wo,
              const float* __restrict__ bo, float* __restrict__ out) {
  const int lane  = threadIdx.x & 31;
  const int wave  = threadIdx.x >> 5;
  const int rbase = blockIdx.x * 64 + wave * 16;
  const int cbase = blockIdx.y * 64;

  v8f acc[4] = {};
  for (int k0 = 0; k0 < E_DIM; k0 += 32) {
    v16bf a = load_frag_rm(A, E_DIM, rbase, k0, lane);
#pragma unroll
    for (int t = 0; t < 4; ++t) {
      v16bf b = load_frag_rm(Wo, E_DIM, cbase + t * 16, k0, lane);
      acc[t] = wmma_bf16(a, b, acc[t]);
    }
  }
  const int n    = lane & 15;
  const int moff = (lane & 16) ? 8 : 0;
#pragma unroll
  for (int t = 0; t < 4; ++t) {
    int c = cbase + t * 16 + n;
    float bc = bo[c];
#pragma unroll
    for (int v = 0; v < 8; ++v) {
      int r = rbase + moff + v;
      out[(size_t)r * E_DIM + c] = acc[t][v] + bc;
    }
  }
}

extern "C" void kernel_launch(void* const* d_in, const int* in_sizes, int n_in,
                              void* d_out, int out_size, void* d_ws, size_t ws_size,
                              hipStream_t stream) {
  (void)in_sizes; (void)n_in; (void)out_size; (void)ws_size;
  const float* query  = (const float*)d_in[0];
  const float* q_proj = (const float*)d_in[1];
  const float* q_bias = (const float*)d_in[2];
  const float* k_proj = (const float*)d_in[3];
  const float* k_bias = (const float*)d_in[4];
  const float* v_proj = (const float*)d_in[5];
  const float* v_bias = (const float*)d_in[6];
  const float* o_proj = (const float*)d_in[7];
  const float* o_bias = (const float*)d_in[8];

  char* ws = (char*)d_ws;
  size_t P = 0;
  const size_t actB = (size_t)M_ROWS * E_DIM * sizeof(bf16_t);   // 8 MB
  const size_t wB   = (size_t)E_DIM * E_DIM * sizeof(bf16_t);    // 2 MB
  const size_t hB   = (size_t)NH * L_SEQ * DH * sizeof(bf16_t);  // 8 MB

  bf16_t* Xbf = (bf16_t*)(ws + P); P += actB;
  bf16_t* Wqb = (bf16_t*)(ws + P); P += wB;
  bf16_t* Wkb = (bf16_t*)(ws + P); P += wB;
  bf16_t* Wvb = (bf16_t*)(ws + P); P += wB;
  bf16_t* Wob = (bf16_t*)(ws + P); P += wB;
  bf16_t* Qh  = (bf16_t*)(ws + P); P += hB;
  bf16_t* Kh  = (bf16_t*)(ws + P); P += hB;
  bf16_t* Vt  = (bf16_t*)(ws + P); P += hB;
  bf16_t* Ao  = (bf16_t*)(ws + P); P += actB;   // total 48 MB

  // f32 -> bf16 conversions
  {
    int n4 = M_ROWS * E_DIM / 4;
    cvt_f32_bf16_v4<<<n4 / 256, 256, 0, stream>>>(query, Xbf, n4);
    int w4 = E_DIM * E_DIM / 4;
    cvt_f32_bf16_v4<<<w4 / 256, 256, 0, stream>>>(q_proj, Wqb, w4);
    cvt_f32_bf16_v4<<<w4 / 256, 256, 0, stream>>>(k_proj, Wkb, w4);
    cvt_f32_bf16_v4<<<w4 / 256, 256, 0, stream>>>(v_proj, Wvb, w4);
    cvt_f32_bf16_v4<<<w4 / 256, 256, 0, stream>>>(o_proj, Wob, w4);
  }

  // Q/K/V projections (+scale, +head-major / transposed-V layouts)
  qkv_gemm<<<dim3(M_ROWS / 64, E_DIM / 64, 3), 128, 0, stream>>>(
      Xbf, Wqb, Wkb, Wvb, q_bias, k_bias, v_bias, Qh, Kh, Vt);

  // causal flash attention
  attn_kernel<<<dim3(NH, L_SEQ / 64), 128, 0, stream>>>(Qh, Kh, Vt, Ao);

  // output projection -> f32 d_out
  out_gemm<<<dim3(M_ROWS / 64, E_DIM / 64), 128, 0, stream>>>(
      Ao, Wob, o_bias, (float*)d_out);
}
